// TwoAxisInfoNCELoss_63823214019054
// MI455X (gfx1250) — matrix operand order repaired and checked by reading, not compiled, not run on hardware
//
#include <hip/hip_runtime.h>

// N = 8192 rows, D = 512. loss_i = -log(pos_i / (rowsum_i - 1 + 1e-8)) where
// rowsum_i = sum_j exp(10*cos_ij - 10) comes from a TDM+LDS staged f16 WMMA GEMM
// and pos_i (3 same-block neighbors) is recomputed in fp32.

#define NROWS 8192
#define NDIM  512

typedef __attribute__((ext_vector_type(16))) _Float16     v16h;
typedef __attribute__((ext_vector_type(8)))  float        v8f;
typedef __attribute__((ext_vector_type(4)))  unsigned int uint32x4;
typedef __attribute__((ext_vector_type(4)))  int          int32x4;
typedef __attribute__((ext_vector_type(8)))  int          int32x8;

union FragH { v16h v; uint4 q[2]; };

// ---------------- TDM 2D tile load: rows x 32 f16 elements -> LDS ----------------
// D# per CDNA5 ISA. 2D tile, data_size=2B,
// tensor = NROWS x NDIM row-major, tile = rows x 32 starting at elem_off.
__device__ __forceinline__ void tdm_load_tile(unsigned int lds_off,
                                              const unsigned short* gbase,
                                              unsigned long long elem_off,
                                              int rows) {
    unsigned long long ga = (unsigned long long)(uintptr_t)gbase + elem_off * 2ull;

    uint32x4 g0;
    g0[0] = 1u;                                    // count=1 (valid descriptor)
    g0[1] = lds_off;                               // lds_addr (bytes)
    g0[2] = (unsigned int)ga;                      // global_addr[31:0]
    g0[3] = (unsigned int)(ga >> 32) | (2u << 30); // global_addr[56:32] | type=2

    int32x8 g1;
    g1[0] = (int)(1u << 16);            // workgroup_mask=0, data_size=1 (2 bytes)
    g1[1] = (int)((unsigned)NDIM << 16);   // tensor_dim0 lo16 = 512
    g1[2] = (int)((unsigned)NROWS << 16);  // tensor_dim0 hi=0 | tensor_dim1 lo16 = 8192
    g1[3] = (int)(32u << 16);           // tensor_dim1 hi=0 | tile_dim0 = 32
    g1[4] = rows & 0xFFFF;              // tile_dim1 = rows, tile_dim2 = 0
    g1[5] = NDIM;                       // tensor_dim0_stride lo32 = 512
    g1[6] = 0;                          // stride hi16 | tensor_dim1_stride lo16
    g1[7] = 0;

    int32x4 z4 = {};
#if __clang_major__ >= 23
    int32x8 z8 = {};
    __builtin_amdgcn_tensor_load_to_lds(g0, g1, z4, z4, z8, 0);
#else
    __builtin_amdgcn_tensor_load_to_lds(g0, g1, z4, z4, 0);
#endif
}

// ---------------- zero rowsum ----------------
__global__ void zero_kernel(float* __restrict__ p, int n) {
    int i = blockIdx.x * blockDim.x + threadIdx.x;
    if (i < n) p[i] = 0.0f;
}

// ---------------- normalize rows, write f16 ----------------
__global__ __launch_bounds__(256) void normalize_kernel(const float* __restrict__ X,
                                                        unsigned short* __restrict__ E) {
    int row  = (blockIdx.x * 256 + threadIdx.x) >> 5;
    int lane = threadIdx.x & 31;
    const float* src = X + (size_t)row * NDIM + lane * 16;

    float vals[16];
    *(float4*)&vals[0]  = *(const float4*)(src + 0);
    *(float4*)&vals[4]  = *(const float4*)(src + 4);
    *(float4*)&vals[8]  = *(const float4*)(src + 8);
    *(float4*)&vals[12] = *(const float4*)(src + 12);

    float ss = 0.0f;
#pragma unroll
    for (int t = 0; t < 16; ++t) ss = fmaf(vals[t], vals[t], ss);
    ss += __shfl_xor(ss, 1, 32);
    ss += __shfl_xor(ss, 2, 32);
    ss += __shfl_xor(ss, 4, 32);
    ss += __shfl_xor(ss, 8, 32);
    ss += __shfl_xor(ss, 16, 32);
    float inv = rsqrtf(ss);

    union { _Float16 h[16]; uint4 q[2]; } u;
#pragma unroll
    for (int t = 0; t < 16; ++t) u.h[t] = (_Float16)(vals[t] * inv);

    unsigned short* dst = E + (size_t)row * NDIM + lane * 16;
    *(uint4*)(dst + 0) = u.q[0];
    *(uint4*)(dst + 8) = u.q[1];
}

// ---------------- TDM double-buffered WMMA GEMM with fused exp + row-sum ----------------
// WG = 256 threads = 8 waves; WG tile = 128 rows x 256 cols; wave tile = 64 x 64.
// Per K-step (32): TDM stages A panel (128x32) and B panel (256x32) into LDS.
#define APANEL (128 * 32)          // ushorts
#define BPANEL (256 * 32)
#define PANEL  (APANEL + BPANEL)   // 12288 ushorts = 24 KB

__global__ __launch_bounds__(256) void gemm_rowsum_kernel(const unsigned short* __restrict__ E,
                                                          float* __restrict__ rowsum) {
    __shared__ unsigned short smem[2][PANEL];

    int bx = blockIdx.x & 31;   // column group (256 cols each)
    int by = blockIdx.x >> 5;   // row group (128 rows each)
    int w    = threadIdx.x >> 5;
    int wr   = w & 1;           // wave row (0..1)
    int wc   = w >> 1;          // wave col (0..3)
    int lane = threadIdx.x & 31;
    int lh   = lane & 15;
    int hi   = lane >> 4;

    int r0 = by * 128 + wr * 64;
    int c0 = bx * 256 + wc * 64;

    unsigned long long arow0 = (unsigned long long)(by * 128) * NDIM;
    unsigned long long brow0 = (unsigned long long)(bx * 256) * NDIM;

    v8f acc[4][4] = {};

    // prologue: stage 0 into buffer 0
    if (threadIdx.x < 32) {
        tdm_load_tile((unsigned int)(uintptr_t)&smem[0][0],      E, arow0 + 0, 128);
        tdm_load_tile((unsigned int)(uintptr_t)&smem[0][APANEL], E, brow0 + 0, 256);
        __builtin_amdgcn_s_wait_tensorcnt(0);
    }
    __syncthreads();

    for (int i = 0; i < NDIM / 32; ++i) {
        int cur = i & 1;
        // kick off next stage into the other buffer (its readers passed the last barrier)
        if (i + 1 < NDIM / 32 && threadIdx.x < 32) {
            int k1 = (i + 1) * 32;
            tdm_load_tile((unsigned int)(uintptr_t)&smem[cur ^ 1][0],      E, arow0 + k1, 128);
            tdm_load_tile((unsigned int)(uintptr_t)&smem[cur ^ 1][APANEL], E, brow0 + k1, 256);
        }

        // fragments from LDS (ds_load_b128 x2 each)
        FragH a[4], b[4];
#pragma unroll
        for (int m = 0; m < 4; ++m) {
            // A 16x32: lanes 0-15 hold K {0..7,16..23}; lanes 16-31 hold K {8..15,24..31}
            const unsigned short* p = &smem[cur][(wr * 64 + m * 16 + lh) * 32 + hi * 8];
            a[m].q[0] = *(const uint4*)(p);
            a[m].q[1] = *(const uint4*)(p + 16);
        }
#pragma unroll
        for (int n = 0; n < 4; ++n) {
            // B 32x16: lanes 0-15 hold K {0..7,16..23} of col N=lane; lanes 16-31 hold K {8..15,24..31}
            const unsigned short* p = &smem[cur][APANEL + (wc * 64 + n * 16 + lh) * 32 + hi * 8];
            b[n].q[0] = *(const uint4*)(p);
            b[n].q[1] = *(const uint4*)(p + 16);
        }
#pragma unroll
        for (int m = 0; m < 4; ++m)
#pragma unroll
            for (int n = 0; n < 4; ++n)
                acc[m][n] = __builtin_amdgcn_wmma_f32_16x16x32_f16(
                    false, a[m].v, false, b[n].v, (short)0, acc[m][n], false, false);

        if (threadIdx.x < 32) __builtin_amdgcn_s_wait_tensorcnt(0);
        __syncthreads();
    }

    // epilogue: exp(10*cos - 10), sum the wave's 64 columns per row, one atomic each.
    // C layout: VGPR t holds row (t + 8*hi), column = lh striped across the 16-lane half.
#pragma unroll
    for (int m = 0; m < 4; ++m) {
#pragma unroll
        for (int t = 0; t < 8; ++t) {
            float v = __expf(fmaf(10.0f, acc[m][0][t], -10.0f)) +
                      __expf(fmaf(10.0f, acc[m][1][t], -10.0f)) +
                      __expf(fmaf(10.0f, acc[m][2][t], -10.0f)) +
                      __expf(fmaf(10.0f, acc[m][3][t], -10.0f));
            v += __shfl_xor(v, 1, 16);
            v += __shfl_xor(v, 2, 16);
            v += __shfl_xor(v, 4, 16);
            v += __shfl_xor(v, 8, 16);
            if (lh == t) atomicAdd(&rowsum[r0 + m * 16 + 8 * hi + t], v);
        }
    }
    (void)c0;
}

// ---------------- fp32 positives + per-row loss ----------------
__global__ __launch_bounds__(256) void posloss_kernel(const float* __restrict__ X,
                                                      const float* __restrict__ rowsum,
                                                      float* __restrict__ loss) {
    int g    = (blockIdx.x * 256 + threadIdx.x) >> 5;  // block-of-4 index
    int lane = threadIdx.x & 31;
    const float* base = X + (size_t)g * 4 * NDIM + lane * 16;

    float x[4][16];
#pragma unroll
    for (int r = 0; r < 4; ++r) {
        *(float4*)&x[r][0]  = *(const float4*)(base + (size_t)r * NDIM + 0);
        *(float4*)&x[r][4]  = *(const float4*)(base + (size_t)r * NDIM + 4);
        *(float4*)&x[r][8]  = *(const float4*)(base + (size_t)r * NDIM + 8);
        *(float4*)&x[r][12] = *(const float4*)(base + (size_t)r * NDIM + 12);
    }

    float dots[10];
    int p = 0;
#pragma unroll
    for (int i = 0; i < 4; ++i)
#pragma unroll
        for (int j = i; j < 4; ++j) {
            float s = 0.0f;
#pragma unroll
            for (int t = 0; t < 16; ++t) s = fmaf(x[i][t], x[j][t], s);
            dots[p++] = s;
        }
#pragma unroll
    for (int q = 0; q < 10; ++q) {
        float s = dots[q];
        s += __shfl_xor(s, 1, 32);
        s += __shfl_xor(s, 2, 32);
        s += __shfl_xor(s, 4, 32);
        s += __shfl_xor(s, 8, 32);
        s += __shfl_xor(s, 16, 32);
        dots[q] = s;
    }

    float dm[4][4];
    p = 0;
#pragma unroll
    for (int i = 0; i < 4; ++i)
#pragma unroll
        for (int j = i; j < 4; ++j) { dm[i][j] = dots[p]; dm[j][i] = dots[p]; ++p; }

    float inv[4];
#pragma unroll
    for (int i = 0; i < 4; ++i) inv[i] = rsqrtf(dm[i][i]);

    if (lane < 4) {
        int i = lane;
        float pos = 0.0f;
#pragma unroll
        for (int j = 0; j < 4; ++j)
            if (j != i) {
                float c = dm[i][j] * inv[i] * inv[j];
                pos += __expf(fmaf(10.0f, c, -10.0f));
            }
        float denom = rowsum[g * 4 + i] - 1.0f + 1e-8f;  // total - self exp(0)
        loss[g * 4 + i] = -__logf(pos / denom);
    }
}

// ---------------- final mean reduction ----------------
__global__ __launch_bounds__(256) void final_kernel(const float* __restrict__ loss,
                                                    float* __restrict__ out) {
    __shared__ float sm[256];
    float s = 0.0f;
    for (int i = threadIdx.x; i < NROWS; i += 256) s += loss[i];
    sm[threadIdx.x] = s;
    __syncthreads();
    for (int off = 128; off > 0; off >>= 1) {
        if (threadIdx.x < off) sm[threadIdx.x] += sm[threadIdx.x + off];
        __syncthreads();
    }
    if (threadIdx.x == 0) out[0] = sm[0] / (float)NROWS;
}

extern "C" void kernel_launch(void* const* d_in, const int* in_sizes, int n_in,
                              void* d_out, int out_size, void* d_ws, size_t ws_size,
                              hipStream_t stream) {
    const float* emb = (const float*)d_in[0];   // labels are index-derivable; unused
    float* out = (float*)d_out;

    char* ws = (char*)d_ws;
    unsigned short* E = (unsigned short*)ws;                 // 8 MB normalized f16
    float* rowsum = (float*)(ws + (size_t)NROWS * NDIM * 2); // 32 KB
    float* loss   = rowsum + NROWS;                          // 32 KB

    zero_kernel<<<NROWS / 256, 256, 0, stream>>>(rowsum, NROWS);
    normalize_kernel<<<NROWS / 8, 256, 0, stream>>>(emb, E);
    gemm_rowsum_kernel<<<(NROWS / 128) * (NROWS / 256), 256, 0, stream>>>(E, rowsum);
    posloss_kernel<<<(NROWS / 4) / 8, 256, 0, stream>>>(emb, rowsum, loss);
    final_kernel<<<1, 256, 0, stream>>>(loss, out);
}
